// GaussianUpsampling_20486994002737
// MI455X (gfx1250) — matrix operand order repaired and checked by reading, not compiled, not run on hardware
//
#include <hip/hip_runtime.h>
#include <hip/hip_bf16.h>

#define BB 16
#define SS 512
#define HH 256
#define TT 4096

typedef __attribute__((ext_vector_type(16))) __bf16 v16bf;
typedef __attribute__((ext_vector_type(8)))  float  v8f;
typedef int gv4i __attribute__((vector_size(16)));

union Frag16 { v16bf v; unsigned short u[16]; uint4 q[2]; };

__device__ __forceinline__ unsigned short f32_bf16(float f) {
    return __builtin_bit_cast(unsigned short, (__bf16)f);   // native RNE convert
}

#if defined(__has_builtin)
#if __has_builtin(__builtin_amdgcn_global_load_async_to_lds_b128) && \
    __has_builtin(__builtin_amdgcn_s_wait_asynccnt)
#define USE_ASYNC_LDS 1
#endif
#endif
#ifndef USE_ASYNC_LDS
#define USE_ASYNC_LDS 0
#endif

#if USE_ASYNC_LDS
// signature: (v4i AS1* global src, v4i AS3* lds dst, imm offset, imm cpol);
// offset applies to BOTH global src and LDS dst (ISA 08_async_tensor §4.4)
#define ASYNC_B128(gptr, ldsoff, imm)                                          \
    __builtin_amdgcn_global_load_async_to_lds_b128(                            \
        (__attribute__((address_space(1))) gv4i*)(size_t)(gptr),               \
        (__attribute__((address_space(3))) gv4i*)(unsigned)(ldsoff), imm, 0)
#endif

// ---------------- trivial outputs: mel_mask, mel_len, max_mel_len ----------------
__global__ void gu_misc_kernel(const unsigned char* mel_mask, const int* mel_len,
                               const int* max_mel_len,
                               float* mmaskO, float* mlenO, float* mmaxO) {
    int idx = blockIdx.x * blockDim.x + threadIdx.x;
    if (idx < BB * TT) mmaskO[idx] = mel_mask[idx] ? 1.f : 0.f;
    if (idx < BB)      mlenO[idx]  = (float)mel_len[idx];
    if (idx == 0)      mmaxO[0]    = (float)max_mel_len[0];
}

// ---------------- centers: c = cumsum(d) - 0.5*d ; also emit d as f32 ----------------
__global__ __launch_bounds__(SS) void gu_scan_kernel(const int* d_target,
                                                     float* cArr, float* dO) {
    __shared__ float sd[SS];
    int b = blockIdx.x, s = threadIdx.x;
    float dv = (float)d_target[b * SS + s];
    sd[s] = dv;
    __syncthreads();
    for (int off = 1; off < SS; off <<= 1) {
        float add = (s >= off) ? sd[s - off] : 0.f;
        __syncthreads();
        sd[s] += add;
        __syncthreads();
    }
    cArr[b * SS + s] = sd[s] - 0.5f * dv;
    dO[b * SS + s]   = dv;
}

// ---------------- conv embeds, V (bf16, B-fragment swizzled), range predictor ----------------
// Vsw fragment layout (16-bit B 32x16, contiguous k order): frag = (b*16 + s/32)*16 + h/16,
// lane L = (h%16) + 16*((s%32)/16), elem e = s%16, index = frag*512 + L*16 + e.
__global__ __launch_bounds__(HH) void gu_embed_kernel(
    const float* __restrict__ enc, const float* __restrict__ pt,
    const float* __restrict__ et,  const int*   __restrict__ dt,
    const unsigned char* __restrict__ smask,
    const float* Wp, const float* bp, const float* We, const float* be,
    const float* Wd, const float* bd, const float* Wr, const float* br,
    unsigned short* __restrict__ Vsw, float* __restrict__ invArr,
    float* __restrict__ ltArr)
{
    int bs = blockIdx.x;            // b*SS + s
    int b  = bs >> 9;
    int s  = bs & (SS - 1);
    int h  = threadIdx.x;

    float pm = (s > 0)      ? pt[bs - 1] : 0.f;
    float pc = pt[bs];
    float pp = (s < SS - 1) ? pt[bs + 1] : 0.f;
    float em = (s > 0)      ? et[bs - 1] : 0.f;
    float ec = et[bs];
    float ep = (s < SS - 1) ? et[bs + 1] : 0.f;
    float dm = (s > 0)      ? (float)dt[bs - 1] : 0.f;
    float dc = (float)dt[bs];
    float dp = (s < SS - 1) ? (float)dt[bs + 1] : 0.f;

    float pe = fmaf(Wp[h*3], pm, fmaf(Wp[h*3+1], pc, fmaf(Wp[h*3+2], pp, bp[h])));
    float ee = fmaf(We[h*3], em, fmaf(We[h*3+1], ec, fmaf(We[h*3+2], ep, be[h])));
    float de = fmaf(Wd[h*3], dm, fmaf(Wd[h*3+1], dc, fmaf(Wd[h*3+2], dp, bd[h])));

    float V = pe + ee + enc[(size_t)bs * HH + h];

    int kt = s >> 5, so = s & 31;
    int L  = (h & 15) + ((so >> 4) << 4);
    int e  = so & 15;
    size_t frag = (size_t)(b * 16 + kt) * 16 + (h >> 4);
    Vsw[frag * 512 + L * 16 + e] = f32_bf16(V);

    __shared__ float red[HH];
    red[h] = (V + de) * Wr[h];
    __syncthreads();
    for (int o = HH / 2; o > 0; o >>= 1) {
        if (h < o) red[h] += red[h + o];
        __syncthreads();
    }
    if (h == 0) {
        float r  = red[0] + br[0];
        float sp = (r > 20.f) ? r : log1pf(__expf(r));   // softplus
        if (smask[bs]) {
            invArr[bs] = 0.f;
            ltArr[bs]  = -__builtin_inff();              // exp -> 0 for masked tokens
        } else {
            invArr[bs] = 1.f / sp;
            ltArr[bs]  = -__logf(sp) - 0.9189385332046727f;  // -log s - 0.5*log(2pi)
        }
    }
}

// ---------------- attn: two passes over S, one thread per t ----------------
__global__ __launch_bounds__(256) void gu_attn_kernel(
    const float* __restrict__ cArr, const float* __restrict__ invArr,
    const float* __restrict__ ltArr, float* __restrict__ attnO)
{
    __shared__ float cS[SS], iS[SS], lS[SS];
    int b   = blockIdx.x >> 4;          // 16 t-chunks of 256 per batch
    int t0  = (blockIdx.x & 15) * 256;
    int tid = threadIdx.x;
    for (int i = tid; i < SS; i += 256) {
        cS[i] = cArr[b*SS + i]; iS[i] = invArr[b*SS + i]; lS[i] = ltArr[b*SS + i];
    }
    __syncthreads();

    float tf  = (float)(t0 + tid + 1);  // t = 1..T
    float sum = 0.f;
    for (int s = 0; s < SS; ++s) {
        float z = (tf - cS[s]) * iS[s];
        sum += __expf(fmaf(-0.5f * z, z, lS[s]));
    }
    float rn = 1.f / (sum + 1e-8f);
    for (int s = 0; s < SS; ++s) {
        float z = (tf - cS[s]) * iS[s];
        float w = __expf(fmaf(-0.5f * z, z, lS[s]));
        attnO[(size_t)(b*SS + s) * TT + t0 + tid] = w * rn;   // coalesced over tid
    }
}

// ---------------- einsum bst,bsh->bth : bf16 WMMA, double-buffered LDS pipeline ----------------
// Block: 64(t) x 128(h) tile, 256 threads = 8 waves (4 t-tiles x 2 h-strips of 64).
// B fragments staged with GLOBAL_LOAD_ASYNC_TO_LDS_B128 (ASYNCcnt) when available.
__global__ __launch_bounds__(256) void gu_gemm_kernel(
    const float* __restrict__ attn, const unsigned short* __restrict__ Vsw,
    float* __restrict__ outO)
{
    __shared__ __align__(16) unsigned short ldsA[2][4 * 512];   // 2 x 4 KB
    __shared__ __align__(16) unsigned short ldsB[2][8 * 512];   // 2 x 8 KB

    const int NK = SS / 32;

    int hb = blockIdx.x & 1;            // h-block of 128
    int tb = (blockIdx.x >> 1) & 63;    // t-block of 64
    int b  = blockIdx.x >> 7;
    int t_block = tb * 64;

    int tid  = threadIdx.x;
    int wave = tid >> 5;
    int lane = tid & 31;
    int wtl  = wave & 3;                // wave's t-tile (0..3)
    int hs   = wave >> 2;               // wave's h-strip (0..1)
    int laneHalf = lane >> 4;
    int lane16   = lane & 15;

    const float* attnB    = attn + (size_t)b * SS * TT;
    const char*  VswBytes = (const char*)(Vsw + ((size_t)b * 16 * 16 + hb * 8) * 512);

    // per-thread fixed staging coordinates (two 1024-element slices)
    int so0 = tid >> 4,        to40 = (tid & 15) * 4;
    int so1 = so0 + 16,        to41 = to40;
    int e0 = so0 & 15, Lb0 = (so0 >> 4) << 4;
    int e1 = so1 & 15, Lb1 = (so1 >> 4) << 4;

    float4 pa0, pa1;
    pa0 = *(const float4*)&attnB[(size_t)so0 * TT + t_block + to40];
    pa1 = *(const float4*)&attnB[(size_t)so1 * TT + t_block + to41];

#if USE_ASYNC_LDS
    unsigned ldsBoff[2] = { (unsigned)(size_t)&ldsB[0][0] + tid * 16,
                            (unsigned)(size_t)&ldsB[1][0] + tid * 16 };
    {   // B prefetch for kt=0 into buffer 0 (each thread copies 2 x 16B)
        const char* src = VswBytes + tid * 16;
        ASYNC_B128(src, ldsBoff[0], 0);
        ASYNC_B128(src, ldsBoff[0], 4096);
    }
#else
    const uint4* VswB = (const uint4*)VswBytes;
    uint4 pb0 = VswB[tid], pb1 = VswB[tid + 256];
#endif

    v8f acc[4] = {};
    for (int kt = 0; kt < NK; ++kt) {
        int buf = kt & 1;
        // ---- stage A (f32 -> bf16, fragment order) into LDS buffer ----
        {
            unsigned short* A = ldsA[buf];
            float v0[4] = {pa0.x, pa0.y, pa0.z, pa0.w};
            float v1[4] = {pa1.x, pa1.y, pa1.z, pa1.w};
            #pragma unroll
            for (int j = 0; j < 4; ++j) {
                int L0 = ((to40 + j) & 15) + Lb0;
                int L1 = ((to41 + j) & 15) + Lb1;
                A[((to40 + j) >> 4) * 512 + L0 * 16 + e0] = f32_bf16(v0[j]);
                A[((to41 + j) >> 4) * 512 + L1 * 16 + e1] = f32_bf16(v1[j]);
            }
#if !USE_ASYNC_LDS
            uint4* Bd = (uint4*)ldsB[buf];
            Bd[tid]       = pb0;
            Bd[tid + 256] = pb1;
#endif
        }
        // ---- prefetch next K-step (overlaps with barrier + WMMAs) ----
        if (kt + 1 < NK) {
            int k0n = (kt + 1) * 32;
            pa0 = *(const float4*)&attnB[(size_t)(k0n + so0) * TT + t_block + to40];
            pa1 = *(const float4*)&attnB[(size_t)(k0n + so1) * TT + t_block + to41];
#if USE_ASYNC_LDS
            const char* src = VswBytes + (size_t)(kt + 1) * 16384 + tid * 16;
            ASYNC_B128(src, ldsBoff[buf ^ 1], 0);
            ASYNC_B128(src, ldsBoff[buf ^ 1], 4096);
#else
            const uint4* src = VswB + (size_t)(kt + 1) * 1024;   // 16 frags = 16 KB
            pb0 = src[tid];
            pb1 = src[tid + 256];
#endif
        }
#if USE_ASYNC_LDS
        // async loads complete in order: oldest 2 (this kt's B tile) must be done.
        if (kt + 1 < NK) __builtin_amdgcn_s_wait_asynccnt(2);
        else             __builtin_amdgcn_s_wait_asynccnt(0);
#endif
        __syncthreads();
        // ---- fragments from LDS: A once, all 4 B frags, then 4 back-to-back WMMAs ----
        Frag16 a;
        {
            const uint4* pa = (const uint4*)&ldsA[buf][wtl * 512 + lane * 16];
            a.q[0] = pa[0]; a.q[1] = pa[1];
        }
        Frag16 bf[4];
        #pragma unroll
        for (int n = 0; n < 4; ++n) {
            const uint4* pb = (const uint4*)&ldsB[buf][(hs * 4 + n) * 512 + lane * 16];
            bf[n].q[0] = pb[0]; bf[n].q[1] = pb[1];
        }
        #pragma unroll
        for (int n = 0; n < 4; ++n) {
            acc[n] = __builtin_amdgcn_wmma_f32_16x16x32_bf16(
                false, a.v, false, bf[n].v, (short)0, acc[n], false, false);
        }
        // double-buffered: single barrier per iteration is sufficient
    }

    // ---- store C: VGPR v -> M = v + 8*laneHalf, lane16 -> N ----
    float* outB = outO + (size_t)b * TT * HH;
    int h_base = hb * 128 + hs * 64;
    #pragma unroll
    for (int n = 0; n < 4; ++n) {
        int h = h_base + n * 16 + lane16;
        #pragma unroll
        for (int v = 0; v < 8; ++v) {
            int tm = t_block + wtl * 16 + v + laneHalf * 8;
            outB[(size_t)tm * HH + h] = acc[n][v];
        }
    }
}

extern "C" void kernel_launch(void* const* d_in, const int* in_sizes, int n_in,
                              void* d_out, int out_size, void* d_ws, size_t ws_size,
                              hipStream_t stream)
{
    const float* enc = (const float*)d_in[0];
    const float* pt  = (const float*)d_in[1];
    const float* et  = (const float*)d_in[2];
    const int*   dt  = (const int*)d_in[3];
    const unsigned char* smask = (const unsigned char*)d_in[4];
    const unsigned char* mmask = (const unsigned char*)d_in[5];
    const int* mlen = (const int*)d_in[6];
    const int* mmax = (const int*)d_in[7];
    const float* Wp = (const float*)d_in[8];
    const float* bp = (const float*)d_in[9];
    const float* We = (const float*)d_in[10];
    const float* be = (const float*)d_in[11];
    const float* Wd = (const float*)d_in[12];
    const float* bd = (const float*)d_in[13];
    const float* Wr = (const float*)d_in[14];
    const float* br = (const float*)d_in[15];

    // d_out: out (B,T,H) | attn (B,S,T) | mel_len (B) | max_mel_len (1) |
    //        mel_mask (B,T) | d (B,S)  -- all as f32
    float* outO   = (float*)d_out;
    float* attnO  = outO  + (size_t)BB * TT * HH;
    float* mlenO  = attnO + (size_t)BB * SS * TT;
    float* mmaxO  = mlenO + BB;
    float* mmaskO = mmaxO + 1;
    float* dO     = mmaskO + (size_t)BB * TT;

    // workspace: V bf16 swizzled (B,S,H) | c | 1/s | logterm   (~4.3 MB total)
    char* w = (char*)d_ws;
    unsigned short* Vsw = (unsigned short*)w;
    float* cArr   = (float*)(w + (size_t)BB * SS * HH * 2);
    float* invArr = cArr   + BB * SS;
    float* ltArr  = invArr + BB * SS;

    gu_misc_kernel<<<(BB * TT) / 256, 256, 0, stream>>>(mmask, mlen, mmax,
                                                        mmaskO, mlenO, mmaxO);
    gu_scan_kernel<<<BB, SS, 0, stream>>>(dt, cArr, dO);
    gu_embed_kernel<<<BB * SS, HH, 0, stream>>>(enc, pt, et, dt, smask,
                                                Wp, bp, We, be, Wd, bd, Wr, br,
                                                Vsw, invArr, ltArr);
    gu_attn_kernel<<<BB * (TT / 256), 256, 0, stream>>>(cArr, invArr, ltArr, attnO);
    gu_gemm_kernel<<<BB * 64 * 2, 256, 0, stream>>>(attnO, Vsw, outO);
}
